// CDGS8_17592186044851
// MI455X (gfx1250) — matrix operand order repaired and checked by reading, not compiled, not run on hardware
//
#include <hip/hip_runtime.h>
#include <hip/hip_bf16.h>
#include <math.h>

// ---------------- problem constants ----------------
#define BB   32
#define TT   4096
#define TG   1024
#define DD   256
#define NH   8
#define DH   32
#define NG   64

typedef __attribute__((ext_vector_type(16))) _Float16 v16h;
typedef __attribute__((ext_vector_type(8)))  float    v8f;
typedef __attribute__((ext_vector_type(4)))  float    v4f;

union F16x16 { v16h v; _Float16 h[16]; };
union F32x8  { v8f  v; float    f[8];  };

__device__ __forceinline__ float gelu_exact(float x) {
    return 0.5f * x * (1.0f + erff(x * 0.70710678118654752f));
}
__device__ __forceinline__ float sigmoidf_(float x) {
    return 1.0f / (1.0f + expf(-x));
}

// B-fragment loader (32x16, 16-bit): lane r = column n, lanes 0-15 hold K {0..7,16..23}, lanes 16-31 hold K {8..15,24..31}.
__device__ __forceinline__ void load_bfrag(
    F16x16& bf, const float* __restrict__ Wb, int n0x, int r, int hi, int kc,
    long long w_ns, long long w_ks)
{
    if (w_ks == 1) {
        const float* wb = Wb + (long long)(n0x + r) * w_ns + kc + (hi << 3);
        const v4f b0 = *(const v4f*)(wb + 0);
        const v4f b1 = *(const v4f*)(wb + 4);
        const v4f b2 = *(const v4f*)(wb + 16);
        const v4f b3 = *(const v4f*)(wb + 20);
#pragma unroll
        for (int e = 0; e < 4; ++e) {
            bf.h[e]      = (_Float16)b0[e];
            bf.h[e + 4]  = (_Float16)b1[e];
            bf.h[e + 8]  = (_Float16)b2[e];
            bf.h[e + 12] = (_Float16)b3[e];
        }
    } else {
#pragma unroll
        for (int e = 0; e < 16; ++e) {
            const int kb = (e & 7) + ((e >> 3) << 4) + (hi << 3);
            bf.h[e] = (_Float16)Wb[(long long)(n0x + r) * w_ns + (long long)(kc + kb) * w_ks];
        }
    }
}

// ---------------------------------------------------------------------------
// Generic batched WMMA GEMM tile kernel, register-blocked 16(M) x 32(N).
//   C[m,n] = scale * sum_k A[m,k] * W[n,k]  (+ bias[n]) (+ optional GELU)
// A element address: A + y*a_by + z*a_bz + m*a_rs + k*a_ks
// W element address: W + y*w_by + z*w_bz + n*w_ns + k*w_ks
// C element address: C + y*c_by + z*c_bz + m*c_ls + n
// One wave (32 threads) per 16x32 output block (two 16x16 WMMA tiles sharing
// the A fragment). ntn (total 16-wide N tiles) must be even; K % 32 == 0.
// Fragment layouts per CDNA5 ISA 7.12.2 (16-bit A 16x32, B 32x16, f32 C/D).
// Contiguous-k operands are fetched as 128-bit loads.
// ---------------------------------------------------------------------------
__global__ __launch_bounds__(32) void k_gemm_wmma(
    const float* __restrict__ A, long long a_rs, long long a_ks, long long a_by, long long a_bz,
    const float* __restrict__ W, long long w_ns, long long w_ks, long long w_by, long long w_bz,
    const float* __restrict__ bias,
    float* __restrict__ C, long long c_ls, long long c_by, long long c_bz,
    int ntn, int K, float scale, int act)
{
    const int npairs = ntn >> 1;
    const int tm  = blockIdx.x / npairs;
    const int tnp = blockIdx.x % npairs;
    const int lane = threadIdx.x;
    const int r  = lane & 15;
    const int hi = lane >> 4;

    const float* Ab = A + (long long)blockIdx.y * a_by + (long long)blockIdx.z * a_bz;
    const float* Wb = W + (long long)blockIdx.y * w_by + (long long)blockIdx.z * w_bz;
    float*       Cb = C + (long long)blockIdx.y * c_by + (long long)blockIdx.z * c_bz;

    const int m0 = tm * 16, n0 = tnp * 32;

    F32x8 acc0, acc1;
#pragma unroll
    for (int v = 0; v < 8; ++v) { acc0.f[v] = 0.0f; acc1.f[v] = 0.0f; }

    for (int kc = 0; kc < K; kc += 32) {
        // A 16x32 fragment: lanes 0-15 -> K {0..7,16..23}; lanes 16-31 -> K {8..15,24..31}
        F16x16 af;
        if (a_ks == 1) {
            const float* ab = Ab + (long long)(m0 + r) * a_rs + kc + (hi << 3);
            const v4f a0 = *(const v4f*)(ab + 0);
            const v4f a1 = *(const v4f*)(ab + 4);
            const v4f a2 = *(const v4f*)(ab + 16);
            const v4f a3 = *(const v4f*)(ab + 20);
#pragma unroll
            for (int e = 0; e < 4; ++e) {
                af.h[e]      = (_Float16)a0[e];
                af.h[e + 4]  = (_Float16)a1[e];
                af.h[e + 8]  = (_Float16)a2[e];
                af.h[e + 12] = (_Float16)a3[e];
            }
        } else {
#pragma unroll
            for (int e = 0; e < 16; ++e) {
                const int ka = (e & 7) + ((e >> 3) << 4) + (hi << 3);
                af.h[e] = (_Float16)Ab[(long long)(m0 + r) * a_rs + (long long)(kc + ka) * a_ks];
            }
        }

        F16x16 bf0, bf1;
        load_bfrag(bf0, Wb, n0,      r, hi, kc, w_ns, w_ks);
        load_bfrag(bf1, Wb, n0 + 16, r, hi, kc, w_ns, w_ks);

        acc0.v = __builtin_amdgcn_wmma_f32_16x16x32_f16(
            false, af.v, false, bf0.v, (short)0, acc0.v, false, false);
        acc1.v = __builtin_amdgcn_wmma_f32_16x16x32_f16(
            false, af.v, false, bf1.v, (short)0, acc1.v, false, false);
    }

    const int col0 = n0 + r;
    const int col1 = n0 + 16 + r;
    const float bb0 = bias ? bias[col0] : 0.0f;
    const float bb1 = bias ? bias[col1] : 0.0f;
#pragma unroll
    for (int v = 0; v < 8; ++v) {
        const int row = m0 + v + (hi << 3);
        float v0 = acc0.f[v] * scale + bb0;
        float v1 = acc1.f[v] * scale + bb1;
        if (act == 1) { v0 = gelu_exact(v0); v1 = gelu_exact(v1); }
        Cb[(long long)row * c_ls + col0] = v0;
        Cb[(long long)row * c_ls + col1] = v1;
    }
}

// ---------------- softmax over rows of length TG (1024) ----------------
__global__ __launch_bounds__(256) void k_softmax(float* __restrict__ S)
{
    __shared__ float red[256];
    float* p = S + (long long)blockIdx.x * TG;
    const int t = threadIdx.x;
    float x0 = p[t], x1 = p[t + 256], x2 = p[t + 512], x3 = p[t + 768];
    float m = fmaxf(fmaxf(x0, x1), fmaxf(x2, x3));
    red[t] = m; __syncthreads();
    for (int s = 128; s > 0; s >>= 1) { if (t < s) red[t] = fmaxf(red[t], red[t + s]); __syncthreads(); }
    m = red[0]; __syncthreads();
    float e0 = expf(x0 - m), e1 = expf(x1 - m), e2 = expf(x2 - m), e3 = expf(x3 - m);
    red[t] = e0 + e1 + e2 + e3; __syncthreads();
    for (int s = 128; s > 0; s >>= 1) { if (t < s) red[t] += red[t + s]; __syncthreads(); }
    const float inv = 1.0f / red[0];
    p[t] = e0 * inv; p[t + 256] = e1 * inv; p[t + 512] = e2 * inv; p[t + 768] = e3 * inv;
}

// ------------- x = q + xw ; LayerNorm ; += gelu(meta proj) -------------
__global__ __launch_bounds__(256) void k_ln_meta(
    const float* __restrict__ xw, const float* __restrict__ queries,
    const float* __restrict__ lng, const float* __restrict__ lnb,
    const float* __restrict__ meta, const float* __restrict__ mw, const float* __restrict__ mb,
    float* __restrict__ qs)
{
    __shared__ float red[256];
    const int row = blockIdx.x;        // b*NG + g
    const int b = row >> 6, g = row & 63;
    const int d = threadIdx.x;
    const float x = queries[g * DD + d] + xw[(long long)row * DD + d];
    red[d] = x; __syncthreads();
    for (int s = 128; s > 0; s >>= 1) { if (d < s) red[d] += red[d + s]; __syncthreads(); }
    const float mu = red[0] * (1.0f / DD); __syncthreads();
    const float dv = x - mu;
    red[d] = dv * dv; __syncthreads();
    for (int s = 128; s > 0; s >>= 1) { if (d < s) red[d] += red[d + s]; __syncthreads(); }
    const float var = red[0] * (1.0f / DD);
    float ns = dv * rsqrtf(var + 1e-5f) * lng[d] + lnb[d];
    float mv = mb[d];
#pragma unroll
    for (int j = 0; j < 6; ++j) mv += meta[b * 6 + j] * mw[d * 6 + j];
    qs[(long long)row * DD + d] = ns + gelu_exact(mv);
}

// ------------- params = h1 @ ph_w2^T + ph_b2  (N=10, K=128) -------------
__global__ __launch_bounds__(32) void k_params(
    const float* __restrict__ h1, const float* __restrict__ w2,
    const float* __restrict__ b2, float* __restrict__ params)
{
    const int row = blockIdx.x;
    const int o = threadIdx.x;
    if (o < 10) {
        float a = b2[o];
        const float* hr = h1 + (long long)row * 128;
        const float* wr = w2 + o * 128;
        for (int i = 0; i < 128; ++i) a += hr[i] * wr[i];
        params[(long long)row * 10 + o] = a;
    }
}

// ------------- gf_mean[b,d] = mean_t global_feat[b,d,t] -------------
__global__ __launch_bounds__(256) void k_gfmean(
    const float* __restrict__ gf, float* __restrict__ gfm)
{
    __shared__ float red[256];
    const float* p = gf + (long long)blockIdx.x * TG;
    const int t = threadIdx.x;
    red[t] = p[t] + p[t + 256] + p[t + 512] + p[t + 768];
    __syncthreads();
    for (int s = 128; s > 0; s >>= 1) { if (t < s) red[t] += red[t + s]; __syncthreads(); }
    if (t == 0) gfm[blockIdx.x] = red[0] * (1.0f / TG);
}

// ------------- heart_shift[b] = tanh(gf_mean @ shift_w^T + shift_b)*0.3 -------------
__global__ __launch_bounds__(32) void k_heart(
    const float* __restrict__ gfm, const float* __restrict__ sw,
    const float* __restrict__ sb, float* __restrict__ hs)
{
    const int b = blockIdx.x;
    const int j = threadIdx.x;
    if (j < 3) {
        float s = sb[j];
        const float* g = gfm + b * DD;
        const float* w = sw + j * DD;
        for (int d = 0; d < DD; ++d) s += g[d] * w[d];
        hs[b * 3 + j] = tanhf(s) * 0.3f;
    }
}

// ------------- geometry outputs: amplitude, p_pos, dipole_dir, volume -------------
__global__ __launch_bounds__(256) void k_geom(
    const float* __restrict__ params, const float* __restrict__ p0,
    const float* __restrict__ hs, float* __restrict__ out)
{
    const int idx = blockIdx.x * 256 + threadIdx.x;  // b*NG+g
    if (idx >= BB * NG) return;
    const int b = idx >> 6, g = idx & 63;
    const float* p = params + (long long)idx * 10;
    out[idx] = sigmoidf_(p[9]) + 1e-4f;                        // amplitude
    float dn = sqrtf(p[3] * p[3] + p[4] * p[4] + p[5] * p[5]);
    dn = fmaxf(dn, 1e-6f);
#pragma unroll
    for (int c = 0; c < 3; ++c) {
        out[2048 + idx * 3 + c]  = p0[g * 3 + c] + tanhf(p[c]) * 0.2f + hs[b * 3 + c]; // p_pos
        out[8192 + idx * 3 + c]  = p[3 + c] / dn;                                      // dipole_dir
        out[14336 + idx * 3 + c] = sigmoidf_(p[6 + c]) * 0.5f + 0.001f;                // volume_xyz
    }
}

// ------------- per-batch argmax(amplitude) -> delay[b,g] -------------
__global__ __launch_bounds__(64) void k_delay(
    const float* __restrict__ out, const float* __restrict__ wds,
    float* __restrict__ delay)
{
    __shared__ int sidx;
    const int b = blockIdx.x;
    const int g = threadIdx.x;
    if (g == 0) {
        const float* a = out + b * NG;
        float best = a[0]; int bi = 0;
        for (int i = 1; i < NG; ++i) { if (a[i] > best) { best = a[i]; bi = i; } }
        sidx = bi;
    }
    __syncthreads();
    const int is = sidx;
    const float* pp = out + 2048;
    float dx = pp[(b * NG + g) * 3 + 0] - pp[(b * NG + is) * 3 + 0];
    float dy = pp[(b * NG + g) * 3 + 1] - pp[(b * NG + is) * 3 + 1];
    float dz = pp[(b * NG + g) * 3 + 2] - pp[(b * NG + is) * 3 + 2];
    const float dist = sqrtf(dx * dx + dy * dy + dz * dz);
    const float sp = log1pf(expf(wds[0]));  // softplus
    delay[b * NG + g] = fminf(sp * dist, 0.1f);
}

// ------------- delay-shifted sampling + sigmoid envelope -------------
__global__ __launch_bounds__(256) void k_envelope(
    const float* __restrict__ tl, const float* __restrict__ delay,
    const float* __restrict__ eb, float* __restrict__ out)
{
    const long long idx = (long long)blockIdx.x * 256 + threadIdx.x;
    const int t = (int)(idx & (TT - 1));
    const int row = (int)(idx >> 12);
    const float d = delay[row];
    const float bx = -1.0f + 2.0f * (float)t / (float)(TT - 1);
    const float xp = (bx - d + 1.0f) * 0.5f * (float)(TT - 1);
    const float x0f = floorf(xp);
    const int x0 = (int)x0f, x1 = x0 + 1;
    const float w1 = xp - x0f, w0 = 1.0f - w1;
    const float* r = tl + (long long)row * TT;
    const int c0 = min(max(x0, 0), TT - 1);
    const int c1 = min(max(x1, 0), TT - 1);
    const float s0 = (x0 >= 0 && x0 <= TT - 1) ? r[c0] : 0.0f;
    const float s1 = (x1 >= 0 && x1 <= TT - 1) ? r[c1] : 0.0f;
    out[20480 + idx] = sigmoidf_(w0 * s0 + w1 * s1 + eb[0]);
}

// ---------------- workspace layout (float offsets) ----------------
#define OFF_KH   0LL                    // B*TG*D        = 8388608
#define OFF_VH   8388608LL              // B*TG*D        = 8388608
#define OFF_S    16777216LL             // B*NH*NG*TG    = 16777216 (scores)
#define OFF_TL   OFF_S                  // B*NG*T        = 8388608 (reuses scores)
#define OFF_QH   33554432LL             // NG*D          = 16384
#define OFF_CTX  33570816LL             // B*NG*D        = 524288
#define OFF_XW   34095104LL             // B*NG*D        = 524288
#define OFF_QS   34619392LL             // B*NG*D        = 524288
#define OFF_H1   35143680LL             // B*NG*128      = 262144
#define OFF_PAR  35405824LL             // B*NG*10       = 20480
#define OFF_GFM  35426304LL             // B*D           = 8192
#define OFF_HS   35434496LL             // B*3 (pad 128) = 128
#define OFF_DEL  35434624LL             // B*NG          = 2048
#define OFF_TW   35436672LL             // B*NG*D        = 524288
// total end = 35960960 floats ~ 143.8 MB

extern "C" void kernel_launch(void* const* d_in, const int* in_sizes, int n_in,
                              void* d_out, int out_size, void* d_ws, size_t ws_size,
                              hipStream_t stream) {
    (void)in_sizes; (void)n_in; (void)out_size; (void)ws_size;

    const float* lf      = (const float*)d_in[0];   // (B,D,T)
    const float* gf      = (const float*)d_in[1];   // (B,D,TG)
    const float* meta    = (const float*)d_in[2];   // (B,6)
    const float* queries = (const float*)d_in[3];   // (1,NG,D)
    const float* p0      = (const float*)d_in[4];   // (1,NG,3)
    const float* in_w    = (const float*)d_in[5];   // (3D,D)
    const float* in_b    = (const float*)d_in[6];   // (3D,)
    const float* ow      = (const float*)d_in[7];   // (D,D)
    const float* ob      = (const float*)d_in[8];
    const float* lng     = (const float*)d_in[9];
    const float* lnb     = (const float*)d_in[10];
    const float* mw      = (const float*)d_in[11];  // (D,6)
    const float* mb      = (const float*)d_in[12];
    const float* tww     = (const float*)d_in[13];  // (D,D)
    const float* twb     = (const float*)d_in[14];
    const float* sw      = (const float*)d_in[15];  // (3,D)
    const float* sb      = (const float*)d_in[16];
    const float* pw1     = (const float*)d_in[17];  // (128,D)
    const float* pb1     = (const float*)d_in[18];
    const float* pw2     = (const float*)d_in[19];  // (10,128)
    const float* pb2     = (const float*)d_in[20];
    const float* wds     = (const float*)d_in[21];  // scalar
    const float* eb      = (const float*)d_in[22];  // scalar

    float* w   = (float*)d_ws;
    float* out = (float*)d_out;

    // 1) qh = queries @ wq^T + bq   (M=64,N=256,K=256; batch-independent)
    k_gemm_wmma<<<dim3(4 * 8, 1, 1), 32, 0, stream>>>(
        queries, 256, 1, 0, 0,  in_w, 256, 1, 0, 0,  in_b,
        w + OFF_QH, 256, 0, 0,  16, 256, 1.0f, 0);

    // 2) kh = kv @ wk^T + bk   (per batch M=TG, A[m,k]=gf[b,k,m] -> a_rs=1, a_ks=TG)
    k_gemm_wmma<<<dim3(64 * 8, BB, 1), 32, 0, stream>>>(
        gf, 1, TG, (long long)DD * TG, 0,  in_w + DD * DD, 256, 1, 0, 0,  in_b + DD,
        w + OFF_KH, 256, (long long)TG * DD, 0,  16, 256, 1.0f, 0);

    // 3) vh = kv @ wv^T + bv
    k_gemm_wmma<<<dim3(64 * 8, BB, 1), 32, 0, stream>>>(
        gf, 1, TG, (long long)DD * TG, 0,  in_w + 2 * DD * DD, 256, 1, 0, 0,  in_b + 2 * DD,
        w + OFF_VH, 256, (long long)TG * DD, 0,  16, 256, 1.0f, 0);

    // 4) S[b,h] = qh_h @ kh_h^T / sqrt(DH)   (M=64,N=1024,K=32; y=h, z=b)
    k_gemm_wmma<<<dim3(4 * 32, NH, BB), 32, 0, stream>>>(
        w + OFF_QH, 256, 1, DH, 0,
        w + OFF_KH, 256, 1, DH, (long long)TG * DD,
        nullptr,
        w + OFF_S, TG, (long long)NG * TG, (long long)NH * NG * TG,
        64, 32, 0.17677669529663687f, 0);

    // 5) softmax rows (B*NH*NG rows of TG)
    k_softmax<<<BB * NH * NG, 256, 0, stream>>>(w + OFF_S);

    // 6) ctx[b,:,h] = P @ V_h   (M=64,N=32,K=1024; W[n,k]=vh[b,k,h*32+n] -> w_ns=1,w_ks=256)
    k_gemm_wmma<<<dim3(4 * 1, NH, BB), 32, 0, stream>>>(
        w + OFF_S, TG, 1, (long long)NG * TG, (long long)NH * NG * TG,
        w + OFF_VH, 1, 256, DH, (long long)TG * DD,
        nullptr,
        w + OFF_CTX, 256, DH, (long long)NG * DD,
        2, TG, 1.0f, 0);

    // 7) xw = ctx @ out_proj^T + ob   (M=2048,N=256,K=256)
    k_gemm_wmma<<<dim3(128 * 8, 1, 1), 32, 0, stream>>>(
        w + OFF_CTX, 256, 1, 0, 0,  ow, 256, 1, 0, 0,  ob,
        w + OFF_XW, 256, 0, 0,  16, 256, 1.0f, 0);

    // 8) x = q + xw; LayerNorm; += gelu(meta proj)
    k_ln_meta<<<BB * NG, 256, 0, stream>>>(
        w + OFF_XW, queries, lng, lnb, meta, mw, mb, w + OFF_QS);

    // 9) h1 = gelu(q_static @ ph_w1^T + ph_b1)   (M=2048,N=128,K=256)
    k_gemm_wmma<<<dim3(128 * 4, 1, 1), 32, 0, stream>>>(
        w + OFF_QS, 256, 1, 0, 0,  pw1, 256, 1, 0, 0,  pb1,
        w + OFF_H1, 128, 0, 0,  8, 256, 1.0f, 1);

    // 10) params (N=10)
    k_params<<<BB * NG, 32, 0, stream>>>(w + OFF_H1, pw2, pb2, w + OFF_PAR);

    // 11) global_feat mean over TG
    k_gfmean<<<BB * DD, 256, 0, stream>>>(gf, w + OFF_GFM);

    // 12) heart_shift
    k_heart<<<BB, 32, 0, stream>>>(w + OFF_GFM, sw, sb, w + OFF_HS);

    // 13) amplitude / p_pos / dipole / volume -> d_out
    k_geom<<<8, 256, 0, stream>>>(w + OFF_PAR, p0, w + OFF_HS, out);

    // 14) per-batch argmax + delay
    k_delay<<<BB, 64, 0, stream>>>(out, wds, w + OFF_DEL);

    // 15) tw = q_static @ temporal_w^T + temporal_b
    k_gemm_wmma<<<dim3(128 * 8, 1, 1), 32, 0, stream>>>(
        w + OFF_QS, 256, 1, 0, 0,  tww, 256, 1, 0, 0,  twb,
        w + OFF_TW, 256, 0, 0,  16, 256, 1.0f, 0);

    // 16) temporal_logits = tw @ local_feat / 16   (per b: M=64,N=4096,K=256)
    k_gemm_wmma<<<dim3(4 * 128, BB, 1), 32, 0, stream>>>(
        w + OFF_TW, 256, 1, (long long)NG * DD, 0,
        lf, 1, TT, (long long)DD * TT, 0,
        nullptr,
        w + OFF_TL, TT, (long long)NG * TT, 0,
        256, 256, 0.0625f, 0);

    // 17) delay-shifted interpolation + sigmoid envelope -> d_out
    k_envelope<<<(BB * NG * TT) / 256, 256, 0, stream>>>(
        w + OFF_TL, w + OFF_DEL, eb, out);
}
